// WaveletPatchEmbedding_3599182594828
// MI455X (gfx1250) — matrix operand order repaired and checked by reading, not compiled, not run on hardware
//
#include <hip/hip_runtime.h>
#include <hip/hip_bf16.h>

typedef __attribute__((ext_vector_type(16))) _Float16 v16h;
typedef __attribute__((ext_vector_type(8)))  float    v8f;

#define T_LEN    1024
#define C_CH     4
#define PATCHES  128
#define PLEN     16
#define STRIDE_  8
#define DMODEL   128
#define FIN      64          // PLEN/... actually C_CH * 16 time steps = 64 features
#define NTHREADS 256

// db4 decomposition filters, pre-scaled by 1/sqrt(2) (pywt swt norm=True)
__constant__ float c_lo[8] = {
  (float)(-0.010597401784997278 * 0.7071067811865476),
  (float)( 0.032883011666982945 * 0.7071067811865476),
  (float)( 0.030841381835986965 * 0.7071067811865476),
  (float)(-0.18703481171888114  * 0.7071067811865476),
  (float)(-0.02798376941698385  * 0.7071067811865476),
  (float)( 0.6308807679295904   * 0.7071067811865476),
  (float)( 0.7148465705525415   * 0.7071067811865476),
  (float)( 0.23037781330885523  * 0.7071067811865476),
};
// hi[k] = (-1)^k * dec_lo[7-k], scaled by 1/sqrt(2)
__constant__ float c_hi[8] = {
  (float)( 0.23037781330885523  * 0.7071067811865476),
  (float)(-0.7148465705525415   * 0.7071067811865476),
  (float)( 0.6308807679295904   * 0.7071067811865476),
  (float)( 0.02798376941698385  * 0.7071067811865476),
  (float)(-0.18703481171888114  * 0.7071067811865476),
  (float)(-0.030841381835986965 * 0.7071067811865476),
  (float)( 0.032883011666982945 * 0.7071067811865476),
  (float)( 0.010597401784997278 * 0.7071067811865476),
};

// One a-trous level: 8-tap circular correlation with dilation d.
// src -> cA_dst (low-pass) and cD_dst (high-pass). 1024 points / 256 threads.
__device__ __forceinline__ void swt_level(const float* __restrict__ src,
                                          float* __restrict__ cA_dst,
                                          float* __restrict__ cD_dst,
                                          int d, int tid)
{
#pragma unroll
  for (int r = 0; r < 4; ++r) {
    const int t = tid + NTHREADS * r;
    float accA = 0.0f, accD = 0.0f;
#pragma unroll
    for (int k = 0; k < 8; ++k) {
      const float v = src[(t + d * k) & (T_LEN - 1)];
      accA = fmaf(c_lo[k], v, accA);
      accD = fmaf(c_hi[k], v, accD);
    }
    cA_dst[t] = accA;
    cD_dst[t] = accD;
  }
}

__global__ void __launch_bounds__(NTHREADS)
wavelet_patch_embed_kernel(const float* __restrict__ x,
                           const float* __restrict__ W,
                           float* __restrict__ out,
                           int write_n_scalar)
{
  __shared__ float    a0[T_LEN];                 // 4 KB   ping
  __shared__ float    a1[T_LEN];                 // 4 KB   pong
  __shared__ float    coeff[C_CH * T_LEN];       // 16 KB  [cA3, cD3, cD2, cD1]
  __shared__ _Float16 Wh[DMODEL * FIN];          // 16 KB  fp16 weights

  const int bn  = blockIdx.x;                    // flattened (b, n) row: 0..4095
  const int tid = threadIdx.x;

  // ---- stage x row and convert W to f16 in LDS --------------------------
  const float* __restrict__ xr = x + (size_t)bn * T_LEN;
#pragma unroll
  for (int r = 0; r < 4; ++r)
    a0[tid + NTHREADS * r] = xr[tid + NTHREADS * r];
#pragma unroll
  for (int r = 0; r < (DMODEL * FIN) / NTHREADS; ++r) {
    const int i = tid + NTHREADS * r;
    Wh[i] = (_Float16)W[i];
  }
  __syncthreads();

  // ---- SWT: 3 a-trous levels, circular --------------------------------
  // channels: 0=cA3, 1=cD3, 2=cD2, 3=cD1
  swt_level(a0, a1, coeff + 3 * T_LEN, 1, tid);  // level 1 (d=1)
  __syncthreads();
  swt_level(a1, a0, coeff + 2 * T_LEN, 2, tid);  // level 2 (d=2)
  __syncthreads();
  swt_level(a0, coeff + 0 * T_LEN, coeff + 1 * T_LEN, 4, tid); // level 3 (d=4)
  __syncthreads();

  // ---- patch + 1x1-conv GEMM via WMMA ----------------------------------
  // out[p, d] = sum_f feat[p, f] * W[d, f],   f = t_in_patch*4 + channel
  // A: 16 patches x K=64 (two K=32 steps), B: K x 16 dmodel columns.
  const int wave = tid >> 5;        // 0..7  -> patch-row block (pm = wave)
  const int lane = tid & 31;
  const int nl   = lane & 15;
  const int hi   = lane >> 4;       // half-wave selector

  // A fragments in contiguous k order (same as B):
  //   lanes 0-15:  M=lane, K 0..15
  //   lanes 16-31: M=lane-16, K 16..31
  const int patch  = wave * 16 + nl;
  const int kbaseA = hi ? 16 : 0;
  v16h afrag[2];
#pragma unroll
  for (int kk = 0; kk < 2; ++kk) {
#pragma unroll
    for (int i = 0; i < 16; ++i) {
      const int f = kk * 32 + kbaseA + i; // 0..63
      const int t = f >> 2;
      const int c = f & 3;
      int tt = patch * STRIDE_ + t;            // replicate-pad last sample
      if (tt > T_LEN - 1) tt = T_LEN - 1;
      afrag[kk][i] = (_Float16)coeff[c * T_LEN + tt];
    }
  }

  // B fragments in contiguous k order (same as A):
  //   lanes 0-15: N=lane, K 0..15 ; lanes 16-31: N=lane-16, K 16..31
  const int kbaseB = hi ? 16 : 0;
  float* __restrict__ orow = out + (size_t)bn * (PATCHES * DMODEL);

  for (int pn = 0; pn < 8; ++pn) {
    const int dcol = pn * 16 + nl;
    v8f acc = {};
#pragma unroll
    for (int kk = 0; kk < 2; ++kk) {
      // 16 consecutive f16 = 32B, 32B-aligned -> two ds_load_b128
      const v16h bfrag =
          *reinterpret_cast<const v16h*>(&Wh[dcol * FIN + kk * 32 + kbaseB]);
      acc = __builtin_amdgcn_wmma_f32_16x16x32_f16(
          /*neg_a=*/false, afrag[kk], /*neg_b=*/false, bfrag,
          /*c_mod=*/(short)0, acc, /*reuse_a=*/false, /*reuse_b=*/false);
    }
    // C/D layout: VGPR r holds rows M=r (lanes 0-15) and M=r+8 (lanes 16-31)
#pragma unroll
    for (int r = 0; r < 8; ++r) {
      const int m = wave * 16 + r + hi * 8;
      orow[m * DMODEL + dcol] = acc[r];
    }
  }

  // second tuple element: N (=64), written once as the trailing output value
  if (write_n_scalar && bn == 0 && tid == 0)
    out[(size_t)gridDim.x * PATCHES * DMODEL] = 64.0f;
}

extern "C" void kernel_launch(void* const* d_in, const int* in_sizes, int n_in,
                              void* d_out, int out_size, void* d_ws, size_t ws_size,
                              hipStream_t stream)
{
  const float* x = (const float*)d_in[0];     // (B, N, T) fp32
  const float* W = (const float*)d_in[1];     // (D_MODEL, FIN) fp32
  float* out = (float*)d_out;

  const int BN = in_sizes[0] / T_LEN;         // B*N = 4096
  const long long embed_elems = (long long)BN * PATCHES * DMODEL;
  const int write_n = ((long long)out_size > embed_elems) ? 1 : 0;

  hipLaunchKernelGGL(wavelet_patch_embed_kernel,
                     dim3(BN), dim3(NTHREADS), 0, stream,
                     x, W, out, write_n);
}
